// BinaryAssociativeMemory_42047729828371
// MI455X (gfx1250) — compile-verified, matrix operand order checked by reading, not run on hardware
//
#include <hip/hip_runtime.h>

// ---------------- CDNA5 / gfx1250 types ----------------
typedef __attribute__((ext_vector_type(16))) __bf16 v16bf;
typedef __attribute__((ext_vector_type(8)))  float  v8f;
typedef __attribute__((ext_vector_type(8)))  unsigned short ush8;
typedef __attribute__((ext_vector_type(4)))  float  f32x4;
typedef __attribute__((ext_vector_type(4)))  unsigned short u16x4;

#define DEV __device__ __forceinline__

// Problem constants (match reference)
#define BB 4
#define TT 4096
#define DDIM 2048
#define HH 16
#define DH 128
#define CC 128
#define NC 32
#define MM (BB*TT)          // 16384 rows
#define BK 64               // K tile staged in LDS
#define QK_SCALE 0.08838834764831843f

// round-to-nearest-even f32 -> bf16 (as raw u16)
static DEV unsigned short f2bf(float f) {
  unsigned u = __builtin_bit_cast(unsigned, f);
  u += 0x7fffu + ((u >> 16) & 1u);
  return (unsigned short)(u >> 16);
}

// 16-element bf16 WMMA fragment from a row-contiguous bf16 buffer.
// CDNA5 16-bit operand layout: lanes 0-15 hold K={0..7,16..23}, lanes 16-31
// hold K={8..15,24..31}: two 16B loads at +kk and +kk+16, kk=(lane<16?0:8).
static DEV v16bf load_frag(const unsigned short* p, int kk) {
  union { ush8 h[2]; v16bf v; } u;
  u.h[0] = *(const ush8*)(p + kk);
  u.h[1] = *(const ush8*)(p + kk + 16);
  return u.v;
}

static DEV v8f wmma_bf16(v16bf a, v16bf b, v8f c) {
  return __builtin_amdgcn_wmma_f32_16x16x32_bf16(false, a, false, b,
                                                 (short)0, c, false, false);
}

// ---- gfx1250 async global->LDS DMA (tracked by ASYNCcnt) ----
static DEV void async_g2l_b128(unsigned lds_off, const unsigned short* gaddr) {
#if defined(__HIP_DEVICE_COMPILE__)
  asm volatile("global_load_async_to_lds_b128 %0, %1, off"
               :: "v"(lds_off), "v"(gaddr)
               : "memory");
#endif
}
static DEV void wait_async0() {
#if defined(__HIP_DEVICE_COMPILE__)
  asm volatile("s_wait_asynccnt 0x0" ::: "memory");
#endif
}
static DEV unsigned lds_off_of(const void* p) {
  return (unsigned)(unsigned long long)p;   // low 32 bits = LDS byte offset
}

// Stage a 128 x BK bf16 tile (rows with constant global stride) into LDS.
static DEV void stage_strided(const unsigned short* gbase, long gstride,
                              unsigned lds_base, int tid) {
  const int r = tid >> 1, half = tid & 1;
  const unsigned short* g = gbase + (long)r * gstride + half * 32;
  const unsigned l = lds_base + (unsigned)((r * BK + half * 32) * 2);
#pragma unroll
  for (int i = 0; i < 4; ++i)
    async_g2l_b128(l + i * 16, g + i * 8);
}

// ---------------- elementwise f32 -> bf16 cast (x4 vectorized) ----------------
__global__ void cast_f32_bf16(const float* __restrict__ in,
                              unsigned short* __restrict__ out, int n4) {
  int i = blockIdx.x * blockDim.x + threadIdx.x;
  int stride = gridDim.x * blockDim.x;
  for (; i < n4; i += stride) {
    f32x4 v = *(const f32x4*)(in + 4 * (long)i);
    u16x4 o;
#pragma unroll
    for (int j = 0; j < 4; ++j) o[j] = f2bf(v[j]);
    *(u16x4*)(out + 4 * (long)i) = o;
  }
}

// Shared inner step: one 32-wide K slice of a 128x128 tile for one wave.
// All 18 ds_loads are issued before the first WMMA consumes them; with a
// relaxed occupancy target the fragments stay in distinct registers.
static DEV void mma_step_128(const unsigned short* Al, const unsigned short* Bl,
                             int row_stride, int w, int lo, int kk, int koff,
                             v8f acc[8]) {
  v16bf a = load_frag(Al + (w * 16 + lo) * row_stride + koff, kk);
  v16bf bfrag[8];
#pragma unroll
  for (int nt = 0; nt < 8; ++nt)
    bfrag[nt] = load_frag(Bl + (nt * 16 + lo) * row_stride + koff, kk);
#pragma unroll
  for (int nt = 0; nt < 8; ++nt)
    acc[nt] = wmma_bf16(a, bfrag[nt], acc[nt]);
}

// ---------------- QKV projection GEMM (NT): y = x @ W^T ----------------
__global__ void __launch_bounds__(256)
__attribute__((amdgpu_waves_per_eu(2))) qkv_gemm(
    const unsigned short* __restrict__ xb,
    const unsigned short* __restrict__ wq,
    const unsigned short* __restrict__ wk,
    const unsigned short* __restrict__ wv,
    unsigned short* __restrict__ qo,
    unsigned short* __restrict__ ko,
    unsigned short* __restrict__ vo) {
  __shared__ unsigned short Ab[2][128 * BK];
  __shared__ unsigned short Bt[2][128 * BK];

  const int tid  = threadIdx.x;
  const int w    = tid >> 5;
  const int lane = tid & 31;
  const int lo   = lane & 15;
  const int hi   = lane >> 4;
  const int kk   = hi ? 8 : 0;

  const int mode = blockIdx.z;
  const unsigned short* wb = (mode == 0) ? wq : ((mode == 1) ? wk : wv);
  unsigned short* ob       = (mode == 0) ? qo : ((mode == 1) ? ko : vo);

  const int mblk = blockIdx.x * 128;
  const int h    = blockIdx.y;
  const unsigned short* agbase = xb + (long)mblk * DDIM;
  const unsigned short* bgbase = wb + (long)(h * 128) * DDIM;

  const unsigned aoff[2] = { lds_off_of(Ab[0]), lds_off_of(Ab[1]) };
  const unsigned boff[2] = { lds_off_of(Bt[0]), lds_off_of(Bt[1]) };

  stage_strided(agbase, DDIM, aoff[0], tid);
  stage_strided(bgbase, DDIM, boff[0], tid);

  v8f acc[8] = {};
  const int NKB = DDIM / BK;                 // 32
  for (int kb = 0; kb < NKB; ++kb) {
    wait_async0();
    __syncthreads();
    if (kb + 1 < NKB) {
      const int nk0 = (kb + 1) * BK;
      stage_strided(agbase + nk0, DDIM, aoff[(kb + 1) & 1], tid);
      stage_strided(bgbase + nk0, DDIM, boff[(kb + 1) & 1], tid);
    }
    const unsigned short* Al = Ab[kb & 1];
    const unsigned short* Bl = Bt[kb & 1];
#pragma unroll
    for (int kt = 0; kt < 2; ++kt)
      mma_step_128(Al, Bl, BK, w, lo, kk, kt * 32, acc);
    __syncthreads();
  }

#pragma unroll
  for (int nt = 0; nt < 8; ++nt) {
#pragma unroll
    for (int r = 0; r < 8; ++r) {
      const int m  = mblk + w * 16 + r + 8 * hi;
      const int b4 = m >> 12;
      const int t  = m & (TT - 1);
      const int dd = nt * 16 + lo;
      const long dst = ((long)(b4 * HH + h) * TT + t) * DH + dd;
      const float v = acc[nt][r];
      unsigned short outv;
      if (mode == 0) outv = f2bf(v * QK_SCALE);
      else outv = (v >= 0.f) ? (unsigned short)0x3F80u
                             : (unsigned short)0xBF80u;
      ob[dst] = outv;
    }
  }
}

// ---------------- chunked linear attention scan ----------------
__global__ void __launch_bounds__(256)
__attribute__((amdgpu_waves_per_eu(2))) bam_attn(
    const unsigned short* __restrict__ qb,
    const unsigned short* __restrict__ kb,
    const unsigned short* __restrict__ vb,
    unsigned short* __restrict__ ob,
    float* __restrict__ state_out) {
  __shared__ unsigned short stateT[DH * DH];   // stateT[e*128 + k]
  __shared__ unsigned short kcT[DH * CC];      // kcT[dd*128 + c]
  __shared__ unsigned short vcT[DH * CC];      // vcT[e*128 + c]
  __shared__ unsigned short sS[8 * 16 * CC];   // per-wave bf16 score strips

  const int tid  = threadIdx.x;
  const int w    = tid >> 5;
  const int lane = tid & 31;
  const int lo   = lane & 15;
  const int hi   = lane >> 4;
  const int kk   = hi ? 8 : 0;

  const int bh = blockIdx.x;
  const unsigned short* qh = qb + (long)bh * TT * DH;
  const unsigned short* kh = kb + (long)bh * TT * DH;
  const unsigned short* vh = vb + (long)bh * TT * DH;
  unsigned short* oh = ob + (long)bh * TT * DH;

  for (int i = tid; i < DH * DH; i += 256) stateT[i] = 0;
  v8f st[8] = {};
  __syncthreads();

  for (int ci = 0; ci < NC; ++ci) {
    const int c0 = ci * CC;

    for (int i = tid; i < CC * DH; i += 256) {
      const int c = i >> 7, dd = i & 127;
      kcT[dd * CC + c] = kh[(long)(c0 + c) * DH + dd];
      vcT[dd * CC + c] = vh[(long)(c0 + c) * DH + dd];
    }
    __syncthreads();

    for (int i = lane; i < 16 * CC; i += 32) sS[w * 16 * CC + i] = 0;

    // ---- phase A: out = qc @ state + tril(qc @ kc^T) @ vc ----
    v8f occ[8] = {};
    v16bf qa[4];
#pragma unroll
    for (int kt = 0; kt < 4; ++kt)
      qa[kt] = load_frag(qh + (long)(c0 + w * 16 + lo) * DH + kt * 32, kk);

    // cross: batch B fragments in groups of 4
#pragma unroll
    for (int kt = 0; kt < 4; ++kt) {
#pragma unroll
      for (int g = 0; g < 2; ++g) {
        v16bf bfrag[4];
#pragma unroll
        for (int j = 0; j < 4; ++j)
          bfrag[j] = load_frag(stateT + ((g * 4 + j) * 16 + lo) * DH + kt * 32, kk);
#pragma unroll
        for (int j = 0; j < 4; ++j)
          occ[g * 4 + j] = wmma_bf16(qa[kt], bfrag[j], occ[g * 4 + j]);
      }
    }

    // scores: batch the 4 global k fragments ahead of the chained WMMAs
    for (int jt = 0; jt <= w; ++jt) {
      v16bf kfr[4];
#pragma unroll
      for (int kt = 0; kt < 4; ++kt)
        kfr[kt] = load_frag(kh + (long)(c0 + jt * 16 + lo) * DH + kt * 32, kk);
      v8f sacc = {};
#pragma unroll
      for (int kt = 0; kt < 4; ++kt)
        sacc = wmma_bf16(qa[kt], kfr[kt], sacc);
#pragma unroll
      for (int r = 0; r < 8; ++r) {
        const int i_loc = r + 8 * hi;
        const bool keep = (jt < w) || (lo <= i_loc);
        sS[w * 16 * CC + i_loc * CC + jt * 16 + lo] = f2bf(keep ? sacc[r] : 0.f);
      }
    }

    // intra: occ += scores @ vc
    const int nblk = (16 * (w + 1) + 31) >> 5;
    for (int kb2 = 0; kb2 < nblk; ++kb2) {
      v16bf sa = load_frag(sS + w * 16 * CC + lo * CC + kb2 * 32, kk);
#pragma unroll
      for (int g = 0; g < 2; ++g) {
        v16bf bfrag[4];
#pragma unroll
        for (int j = 0; j < 4; ++j)
          bfrag[j] = load_frag(vcT + ((g * 4 + j) * 16 + lo) * CC + kb2 * 32, kk);
#pragma unroll
        for (int j = 0; j < 4; ++j)
          occ[g * 4 + j] = wmma_bf16(sa, bfrag[j], occ[g * 4 + j]);
      }
    }

#pragma unroll
    for (int nt = 0; nt < 8; ++nt) {
#pragma unroll
      for (int r = 0; r < 8; ++r) {
        const int t = c0 + w * 16 + r + 8 * hi;
        oh[(long)t * DH + nt * 16 + lo] = f2bf(occ[nt][r]);
      }
    }

    __syncthreads();   // stateT readers done before update

    // ---- phase B: state += kc^T @ vc ----
#pragma unroll
    for (int kt = 0; kt < 4; ++kt) {
      v16bf ka = load_frag(kcT + (w * 16 + lo) * CC + kt * 32, kk);
#pragma unroll
      for (int g = 0; g < 2; ++g) {
        v16bf bfrag[4];
#pragma unroll
        for (int j = 0; j < 4; ++j)
          bfrag[j] = load_frag(vcT + ((g * 4 + j) * 16 + lo) * CC + kt * 32, kk);
#pragma unroll
        for (int j = 0; j < 4; ++j)
          st[g * 4 + j] = wmma_bf16(ka, bfrag[j], st[g * 4 + j]);
      }
    }
#pragma unroll
    for (int nt = 0; nt < 8; ++nt) {
#pragma unroll
      for (int r = 0; r < 8; ++r) {
        const int ke = w * 16 + r + 8 * hi;
        const int e  = nt * 16 + lo;
        stateT[e * DH + ke] = f2bf(st[nt][r]);
      }
    }
    __syncthreads();
  }

  float* so = state_out + (long)bh * DH * DH;
#pragma unroll
  for (int nt = 0; nt < 8; ++nt) {
#pragma unroll
    for (int r = 0; r < 8; ++r)
      so[(long)(w * 16 + r + 8 * hi) * DH + nt * 16 + lo] = st[nt][r];
  }
}

// ---------------- output projection GEMM: out = attn @ Wo^T ----------------
__global__ void __launch_bounds__(256)
__attribute__((amdgpu_waves_per_eu(2))) out_gemm(
    const unsigned short* __restrict__ ab,
    const unsigned short* __restrict__ wo,
    float* __restrict__ out) {
  __shared__ unsigned short Ab[2][128 * BK];
  __shared__ unsigned short Bt[2][128 * BK];

  const int tid  = threadIdx.x;
  const int w    = tid >> 5;
  const int lane = tid & 31;
  const int lo   = lane & 15;
  const int hi   = lane >> 4;
  const int kk   = hi ? 8 : 0;

  const int mblk = blockIdx.x * 128;
  const int n0   = blockIdx.y * 128;

  const int r2 = tid >> 1, half2 = tid & 1;
  const int mr2 = mblk + r2;
  const int b2  = mr2 >> 12;
  const int t2  = mr2 & (TT - 1);

  const unsigned short* bgbase = wo + (long)n0 * DDIM;
  const unsigned aoff[2] = { lds_off_of(Ab[0]), lds_off_of(Ab[1]) };
  const unsigned boff[2] = { lds_off_of(Bt[0]), lds_off_of(Bt[1]) };

  {
    const unsigned short* g =
        ab + ((long)(b2 * HH + 0) * TT + t2) * DH + half2 * 32;
    const unsigned l = aoff[0] + (unsigned)((r2 * BK + half2 * 32) * 2);
#pragma unroll
    for (int i = 0; i < 4; ++i) async_g2l_b128(l + i * 16, g + i * 8);
  }
  stage_strided(bgbase, DDIM, boff[0], tid);

  v8f acc[8] = {};
  const int NKB = DDIM / BK;
  for (int kb = 0; kb < NKB; ++kb) {
    wait_async0();
    __syncthreads();
    if (kb + 1 < NKB) {
      const int nk0  = (kb + 1) * BK;
      const int h    = nk0 >> 7;
      const int koff = nk0 & 127;
      const unsigned short* g =
          ab + ((long)(b2 * HH + h) * TT + t2) * DH + koff + half2 * 32;
      const unsigned l = aoff[(kb + 1) & 1] +
                         (unsigned)((r2 * BK + half2 * 32) * 2);
#pragma unroll
      for (int i = 0; i < 4; ++i) async_g2l_b128(l + i * 16, g + i * 8);
      stage_strided(bgbase + nk0, DDIM, boff[(kb + 1) & 1], tid);
    }
    const unsigned short* Al = Ab[kb & 1];
    const unsigned short* Bl = Bt[kb & 1];
#pragma unroll
    for (int kt = 0; kt < 2; ++kt)
      mma_step_128(Al, Bl, BK, w, lo, kk, kt * 32, acc);
    __syncthreads();
  }

#pragma unroll
  for (int nt = 0; nt < 8; ++nt) {
#pragma unroll
    for (int r = 0; r < 8; ++r) {
      const int mr = mblk + w * 16 + r + 8 * hi;
      out[(long)mr * DDIM + n0 + nt * 16 + lo] = acc[nt][r];
    }
  }
}

// ---------------- host launcher ----------------
extern "C" void kernel_launch(void* const* d_in, const int* in_sizes, int n_in,
                              void* d_out, int out_size, void* d_ws, size_t ws_size,
                              hipStream_t stream) {
  (void)in_sizes; (void)n_in; (void)out_size; (void)ws_size;

  const float* x  = (const float*)d_in[0];
  const float* Wq = (const float*)d_in[1];
  const float* Wk = (const float*)d_in[2];
  const float* Wv = (const float*)d_in[3];
  const float* Wo = (const float*)d_in[4];

  float* out    = (float*)d_out;
  float* fstate = out + (long)BB * TT * DDIM;

  char* p = (char*)d_ws;
  unsigned short* xb  = (unsigned short*)p;  p += (long)MM * DDIM * 2;
  unsigned short* wqb = (unsigned short*)p;  p += (long)DDIM * DDIM * 2;
  unsigned short* wkb = (unsigned short*)p;  p += (long)DDIM * DDIM * 2;
  unsigned short* wvb = (unsigned short*)p;  p += (long)DDIM * DDIM * 2;
  unsigned short* wob = (unsigned short*)p;  p += (long)DDIM * DDIM * 2;
  unsigned short* qbf = (unsigned short*)p;  p += (long)MM * DDIM * 2;
  unsigned short* kbf = (unsigned short*)p;  p += (long)MM * DDIM * 2;
  unsigned short* vbf = (unsigned short*)p;  p += (long)MM * DDIM * 2;
  unsigned short* abf = (unsigned short*)p;  p += (long)MM * DDIM * 2;

  cast_f32_bf16<<<2048, 256, 0, stream>>>(x,  xb,  MM * DDIM / 4);
  cast_f32_bf16<<<512,  256, 0, stream>>>(Wq, wqb, DDIM * DDIM / 4);
  cast_f32_bf16<<<512,  256, 0, stream>>>(Wk, wkb, DDIM * DDIM / 4);
  cast_f32_bf16<<<512,  256, 0, stream>>>(Wv, wvb, DDIM * DDIM / 4);
  cast_f32_bf16<<<512,  256, 0, stream>>>(Wo, wob, DDIM * DDIM / 4);

  qkv_gemm<<<dim3(MM / 128, DDIM / 128, 3), 256, 0, stream>>>(
      xb, wqb, wkb, wvb, qbf, kbf, vbf);

  bam_attn<<<BB * HH, 256, 0, stream>>>(qbf, kbf, vbf, abf, fstate);

  out_gemm<<<dim3(MM / 128, DDIM / 128), 256, 0, stream>>>(abf, wob, out);
}
